// DecodeLSTM_52939766891158
// MI455X (gfx1250) — matrix operand order repaired and checked by reading, not compiled, not run on hardware
//
#include <hip/hip_runtime.h>
#include <hip/hip_bf16.h>
#include <stdint.h>

// ---------------------------------------------------------------------------
// DecodeLSTM on MI455X (gfx1250): 4 sequential fused GEMM+LSTM-cell kernels
// using v_wmma_f32_16x16x32_bf16 with async-to-LDS staging (ASYNCcnt) when
// available, plus one-time weight prep and a dense head.
// B=8192, T=1, H=1024, L=4.  N = 4H = 4096, K = H = 1024.
// ---------------------------------------------------------------------------

#define Bsz   8192
#define Hsz   1024
#define Nsz   4096   // 4*H
#define Ksz   1024

typedef __attribute__((ext_vector_type(16))) __bf16 v16bf;
typedef __attribute__((ext_vector_type(8)))  __bf16 v8bf;
typedef __attribute__((ext_vector_type(8)))  float  v8f;
typedef __attribute__((ext_vector_type(4)))  int    v4i;

#if defined(__has_builtin)
#  if __has_builtin(__builtin_amdgcn_global_load_async_to_lds_b128)
#    define USE_ASYNC_LDS 1
#  endif
#endif
#ifndef USE_ASYNC_LDS
#  define USE_ASYNC_LDS 0
#endif

// ---- workspace layout (bytes) ----
#define WT_OFF    ((size_t)0)                         // 4 layers * 4096*1024 bf16
#define WT_BYTES  ((size_t)4 * Nsz * Ksz * 2)
#define BIAS_OFF  (WT_OFF + WT_BYTES)                 // 4 * 4096 f32 (gate-interleaved)
#define BIAS_BYTES ((size_t)4 * Nsz * 4)
#define WX0_OFF   (BIAS_OFF + BIAS_BYTES)             // 4096 f32 (gate-interleaved)
#define WX0_BYTES ((size_t)Nsz * 4)
#define HBF0_OFF  (WX0_OFF + WX0_BYTES)               // 8192*1024 bf16 ping
#define HBF_BYTES ((size_t)Bsz * Hsz * 2)
#define HBF1_OFF  (HBF0_OFF + HBF_BYTES)              // pong
#define CB0_OFF   (HBF1_OFF + HBF_BYTES)              // 8192*1024 f32 ping
#define CB_BYTES  ((size_t)Bsz * Hsz * 4)
#define CB1_OFF   (CB0_OFF + CB_BYTES)                // pong

__device__ __forceinline__ float sigf(float x) {
    return 1.0f / (1.0f + __expf(-x));
}
__device__ __forceinline__ float tanh_fast(float x) {
    return 2.0f / (1.0f + __expf(-2.0f * x)) - 1.0f;   // 2*sigmoid(2x)-1
}

#if USE_ASYNC_LDS
typedef __attribute__((address_space(1))) v4i* gv4i_p;   // global int4*
typedef __attribute__((address_space(3))) v4i* lv4i_p;   // LDS int4*

__device__ __forceinline__ void async_b128(const void* g, void* l) {
    __builtin_amdgcn_global_load_async_to_lds_b128((gv4i_p)g, (lv4i_p)l, 0, 0);
}
__device__ __forceinline__ void wait_async0() {
#if __has_builtin(__builtin_amdgcn_s_wait_asynccnt)
    __builtin_amdgcn_s_wait_asynccnt(0);
#else
    asm volatile("s_wait_asynccnt 0x0" ::: "memory");
#endif
}
#endif

// ---------------------------------------------------------------------------
// Weight prep: Wt[l][n][k] = bf16( W[k][col(n)] ), col(n) = (n&3)*H + (n>>2)
// layer 0 uses Wh0 alone; layers 1..3 use Wx[l-1]+Wh[l-1] (x_t == h when T=1).
// 32x32 LDS-tiled transpose.  grid (K/32, N/32, 4), block (32,8).
// ---------------------------------------------------------------------------
__global__ void prep_weights_kernel(const float* __restrict__ Wh0,
                                    const float* __restrict__ Wx,
                                    const float* __restrict__ Wh,
                                    __bf16* __restrict__ Wt) {
    __shared__ float tile[32][33];
    const int k0 = blockIdx.x * 32;
    const int n0 = blockIdx.y * 32;
    const int l  = blockIdx.z;
    const int tx = threadIdx.x, ty0 = threadIdx.y;
#pragma unroll
    for (int i = 0; i < 4; ++i) {
        const int ty = ty0 + i * 8;
        const int k = k0 + ty;
        const int ncol = n0 + tx;
        const int col = (ncol & 3) * Hsz + (ncol >> 2);
        float v;
        if (l == 0) {
            v = Wh0[(size_t)k * Nsz + col];
        } else {
            const size_t o = (size_t)(l - 1) * Ksz * Nsz + (size_t)k * Nsz + col;
            v = Wx[o] + Wh[o];
        }
        tile[ty][tx] = v;
    }
    __syncthreads();
#pragma unroll
    for (int i = 0; i < 4; ++i) {
        const int ty = ty0 + i * 8;
        const int n = n0 + ty;
        const int k = k0 + tx;
        Wt[(size_t)l * Nsz * Ksz + (size_t)n * Ksz + k] = (__bf16)tile[tx][ty];
    }
}

// Bias + Wx0 permutation into gate-interleaved order.  grid(80), block(256).
__global__ void prep_vec_kernel(const float* __restrict__ b0,
                                const float* __restrict__ b,
                                const float* __restrict__ Wx0,
                                float* __restrict__ bias_perm,
                                float* __restrict__ wx0_perm) {
    const int idx = blockIdx.x * 256 + threadIdx.x;
    if (idx < 4 * Nsz) {
        const int l = idx >> 12;
        const int ncol = idx & (Nsz - 1);
        const int col = (ncol & 3) * Hsz + (ncol >> 2);
        bias_perm[idx] = (l == 0) ? b0[col] : b[(l - 1) * Nsz + col];
    } else if (idx < 5 * Nsz) {
        const int ncol = idx - 4 * Nsz;
        const int col = (ncol & 3) * Hsz + (ncol >> 2);
        wx0_perm[ncol] = Wx0[col];
    }
}

// h0 f32 -> bf16.  grid(8192), block(256), 4 elems/thread.
__global__ void convert_h0_kernel(const float* __restrict__ h0,
                                  __bf16* __restrict__ hbf) {
    const size_t i0 = ((size_t)blockIdx.x * 256 + threadIdx.x) * 4;
#pragma unroll
    for (int r = 0; r < 4; ++r) hbf[i0 + r] = (__bf16)h0[i0 + r];
}

// ---------------------------------------------------------------------------
// Fused GEMM (bf16 WMMA, f32 accum) + LSTM cell.
// Block tile 128(M) x 128(N); 8 waves arranged 2(M) x 4(N); each wave owns
// 4 M-tiles x 2 N-tiles of 16x16 -> 8 v_wmma_f32_16x16x32_bf16 per K-step,
// 32 K-steps, double-buffered LDS, async-to-LDS staging waited only AFTER the
// WMMA block of each iteration (latency hidden behind matrix math).
// Gate-interleaved columns: n = 4*j + gate, so each lane-quad {4q..4q+3}
// holds (i,f,g,o) of one hidden unit -> cell update via __shfl.
// ---------------------------------------------------------------------------
template <bool FIRST, bool FINAL>
__global__ __launch_bounds__(256, 2)
void lstm_layer_kernel(const __bf16* __restrict__ A,      // [8192,1024] bf16 (h_prev)
                       const __bf16* __restrict__ Wt,     // [4096,1024] bf16, gate-interleaved N
                       const float*  __restrict__ bias,   // [4096] gate-interleaved
                       const float*  __restrict__ wx0,    // [4096] gate-interleaved (FIRST only)
                       const float*  __restrict__ x,      // [8192] (FIRST only)
                       const float*  __restrict__ c_prev, // [8192,1024] f32
                       float*        __restrict__ c_out,  // [8192,1024] f32
                       float*        __restrict__ h_out_f32,  // FINAL only
                       __bf16*       __restrict__ h_out_bf16) // !FINAL only
{
    __shared__ __bf16 As[2][128][40];   // [m][k], pad 40 to dodge bank conflicts
    __shared__ __bf16 Bs[2][128][40];   // [n][k] (Wt already K-major)
    __shared__ float  xsh[128];

    const int tid    = threadIdx.x;
    const int wid    = tid >> 5;
    const int lane   = tid & 31;
    const int wave_m = wid >> 2;        // 0..1  -> 64 rows each
    const int wave_n = wid & 3;         // 0..3  -> 32 cols each
    const int half   = lane >> 4;       // 0/1
    const int l15    = lane & 15;

    const int mb = blockIdx.x * 128;
    const int nb = blockIdx.y * 128;

    if (FIRST && tid < 128) xsh[tid] = x[mb + tid];

    // per-thread staging slot: 32 bytes of one row of each 128x32 tile
    const int rS = tid >> 1;            // 0..127
    const int cS = (tid & 1) * 16;      // 0 or 16 (bf16 elems)
    const __bf16* gA = A  + (size_t)(mb + rS) * Ksz + cS;
    const __bf16* gB = Wt + (size_t)(nb + rS) * Ksz + cS;

#if USE_ASYNC_LDS
    auto stage = [&](int buf, int kb) {
        async_b128(gA + kb,     &As[buf][rS][cS]);
        async_b128(gA + kb + 8, &As[buf][rS][cS + 8]);
        async_b128(gB + kb,     &Bs[buf][rS][cS]);
        async_b128(gB + kb + 8, &Bs[buf][rS][cS + 8]);
    };
    stage(0, 0);
#else
    uint4 ra0, ra1, rb0, rb1;
    auto fetch = [&](int kb) {
        ra0 = *(const uint4*)(gA + kb);
        ra1 = *(const uint4*)(gA + kb + 8);
        rb0 = *(const uint4*)(gB + kb);
        rb1 = *(const uint4*)(gB + kb + 8);
    };
    auto store = [&](int buf) {
        *(uint4*)&As[buf][rS][cS]     = ra0;
        *(uint4*)&As[buf][rS][cS + 8] = ra1;
        *(uint4*)&Bs[buf][rS][cS]     = rb0;
        *(uint4*)&Bs[buf][rS][cS + 8] = rb1;
    };
    fetch(0);
    store(0);
#endif

    v8f acc[8];                          // [m-tile t][n-tile nt] -> acc[t*2+nt]
#pragma unroll
    for (int t = 0; t < 8; ++t) acc[t] = (v8f){0.f, 0.f, 0.f, 0.f, 0.f, 0.f, 0.f, 0.f};

#if USE_ASYNC_LDS
    wait_async0();
#endif
    __syncthreads();

    for (int kt = 0; kt < 32; ++kt) {
        const int buf = kt & 1;
        if (kt < 31) {
#if USE_ASYNC_LDS
            stage(buf ^ 1, (kt + 1) * 32);     // in flight during this iter's WMMAs
#else
            fetch((kt + 1) * 32);              // regs held across the WMMA block
#endif
            __builtin_prefetch(gA + (kt + 2) * 32, 0, 1);  // global_prefetch_b8
            __builtin_prefetch(gB + (kt + 2) * 32, 0, 1);
        }

        // A fragments: lanes 0-15: K 0..15; lanes 16-31: K 16..31
        const int ka = half * 16;
        v16bf afrag[4];
#pragma unroll
        for (int t = 0; t < 4; ++t) {
            const int row = wave_m * 64 + t * 16 + l15;
            ((v8bf*)&afrag[t])[0] = *(const v8bf*)&As[buf][row][ka];
            ((v8bf*)&afrag[t])[1] = *(const v8bf*)&As[buf][row][ka + 8];
        }
        // B fragments: lanes 0-15: K 0..15; lanes 16-31: K 16..31
        const int kb0 = half * 16;
        v16bf bfrag[2];
#pragma unroll
        for (int nt = 0; nt < 2; ++nt) {
            const int nrow = wave_n * 32 + nt * 16 + l15;
            ((v8bf*)&bfrag[nt])[0] = *(const v8bf*)&Bs[buf][nrow][kb0];
            ((v8bf*)&bfrag[nt])[1] = *(const v8bf*)&Bs[buf][nrow][kb0 + 8];
        }

#pragma unroll
        for (int t = 0; t < 4; ++t)
#pragma unroll
            for (int nt = 0; nt < 2; ++nt)
                acc[t * 2 + nt] = __builtin_amdgcn_wmma_f32_16x16x32_bf16(
                    false, afrag[t], false, bfrag[nt], (short)0, acc[t * 2 + nt],
                    false, false);

        if (kt < 31) {
#if USE_ASYNC_LDS
            wait_async0();                     // own wave's async writes landed
#else
            store(buf ^ 1);                    // waits happen here, after WMMAs
#endif
        }
        __syncthreads();
    }

    // ---- epilogue: LSTM cell via lane-quad gate exchange ----
    const int qb = lane & ~3;                          // quad base lane

#pragma unroll
    for (int nt = 0; nt < 2; ++nt) {
        const int nglob = nb + wave_n * 32 + nt * 16 + l15;
        const float biasl = bias[nglob];
        const float wx0l  = FIRST ? wx0[nglob] : 0.0f;
        const int j = (nb >> 2) + wave_n * 8 + nt * 4 + (l15 >> 2); // hidden unit
#pragma unroll
        for (int t = 0; t < 4; ++t) {
#pragma unroll
            for (int v = 0; v < 8; ++v) {
                float z = acc[t * 2 + nt][v] + biasl;
                const int mloc = wave_m * 64 + t * 16 + half * 8 + v;
                if (FIRST) z += xsh[mloc] * wx0l;
                const float zi = __shfl(z, qb + 0, 32);
                const float zf = __shfl(z, qb + 1, 32);
                const float zg = __shfl(z, qb + 2, 32);
                const float zo = __shfl(z, qb + 3, 32);
                if ((v >> 1) == (lane & 3)) {          // each quad lane owns 2 rows
                    const float ig = sigf(zi);
                    const float fg = sigf(zf);
                    const float gg = tanh_fast(zg);
                    const float og = sigf(zo);
                    const int m = mb + mloc;
                    const size_t o = (size_t)m * Hsz + j;
                    const float cp = c_prev[o];
                    const float cn = fg * cp + ig * gg;
                    const float hn = og * tanh_fast(cn);
                    c_out[o] = cn;
                    if (FINAL) h_out_f32[o] = hn;
                    else       h_out_bf16[o] = (__bf16)hn;
                }
            }
        }
    }
}

// ---------------------------------------------------------------------------
// Dense head: x_pred[m] = dot(h[m,:], Wd) + bd.  1 wave per row, shfl reduce.
// grid(1024), block(256) -> 8 rows/block.
// ---------------------------------------------------------------------------
__global__ void head_kernel(const float* __restrict__ h,
                            const float* __restrict__ Wd,
                            const float* __restrict__ bd,
                            float* __restrict__ xpred) {
    const int wid = threadIdx.x >> 5;
    const int lane = threadIdx.x & 31;
    const int row = blockIdx.x * 8 + wid;
    float s = 0.0f;
    for (int k = lane; k < Hsz; k += 32)
        s = fmaf(h[(size_t)row * Hsz + k], Wd[k], s);
#pragma unroll
    for (int off = 16; off > 0; off >>= 1)
        s += __shfl_xor(s, off, 32);
    if (lane == 0) xpred[row] = s + bd[0];
}

// ---------------------------------------------------------------------------
extern "C" void kernel_launch(void* const* d_in, const int* in_sizes, int n_in,
                              void* d_out, int out_size, void* d_ws, size_t ws_size,
                              hipStream_t stream) {
    const float* x     = (const float*)d_in[0];   // [8192]
    const float* c0    = (const float*)d_in[1];   // [8192,1024]
    const float* h0    = (const float*)d_in[2];   // [8192,1024]
    const float* Wx0   = (const float*)d_in[3];   // [1,4096]
    const float* Wh0   = (const float*)d_in[4];   // [1024,4096]
    const float* b0    = (const float*)d_in[5];   // [4096]
    const float* Wx    = (const float*)d_in[6];   // [3,1024,4096]
    const float* Wh    = (const float*)d_in[7];   // [3,1024,4096]
    const float* b     = (const float*)d_in[8];   // [3,4096]
    const float* Wd    = (const float*)d_in[9];   // [1024]
    const float* bd    = (const float*)d_in[10];  // [1]

    char* ws = (char*)d_ws;
    __bf16* Wt        = (__bf16*)(ws + WT_OFF);
    float*  bias_perm = (float*)(ws + BIAS_OFF);
    float*  wx0_perm  = (float*)(ws + WX0_OFF);
    __bf16* hbf0      = (__bf16*)(ws + HBF0_OFF);
    __bf16* hbf1      = (__bf16*)(ws + HBF1_OFF);
    float*  cb0       = (float*)(ws + CB0_OFF);
    float*  cb1       = (float*)(ws + CB1_OFF);

    float* out_c  = (float*)d_out;                         // [8192,1024]
    float* out_h  = out_c + (size_t)Bsz * Hsz;             // [8192,1024]
    float* out_xp = out_h + (size_t)Bsz * Hsz;             // [8192]

    prep_weights_kernel<<<dim3(Ksz / 32, Nsz / 32, 4), dim3(32, 8), 0, stream>>>(
        Wh0, Wx, Wh, Wt);
    prep_vec_kernel<<<dim3(80), dim3(256), 0, stream>>>(b0, b, Wx0, bias_perm, wx0_perm);
    convert_h0_kernel<<<dim3((Bsz * Hsz) / 1024), dim3(256), 0, stream>>>(h0, hbf0);

    const dim3 ggrid(Bsz / 128, Nsz / 128);
    const dim3 gblk(256);
    const size_t WTL = (size_t)Nsz * Ksz;

    // layer 0: A = bf16(h0), c_prev = c0; rank-1 x*Wx0 folded into epilogue
    lstm_layer_kernel<true, false><<<ggrid, gblk, 0, stream>>>(
        hbf0, Wt + 0 * WTL, bias_perm + 0 * Nsz, wx0_perm, x,
        c0, cb0, nullptr, hbf1);
    // layer 1
    lstm_layer_kernel<false, false><<<ggrid, gblk, 0, stream>>>(
        hbf1, Wt + 1 * WTL, bias_perm + 1 * Nsz, nullptr, nullptr,
        cb0, cb1, nullptr, hbf0);
    // layer 2
    lstm_layer_kernel<false, false><<<ggrid, gblk, 0, stream>>>(
        hbf0, Wt + 2 * WTL, bias_perm + 2 * Nsz, nullptr, nullptr,
        cb1, cb0, nullptr, hbf1);
    // layer 3 (final): c,h straight into d_out as f32
    lstm_layer_kernel<false, true><<<ggrid, gblk, 0, stream>>>(
        hbf1, Wt + 3 * WTL, bias_perm + 3 * Nsz, nullptr, nullptr,
        cb0, out_c, out_h, nullptr);

    head_kernel<<<dim3(Bsz / 8), dim3(256), 0, stream>>>(out_h, Wd, bd, out_xp);

    (void)in_sizes; (void)n_in; (void)out_size; (void)ws_size;
}
